// PhysicsInformedGNN_48086453846272
// MI455X (gfx1250) — matrix operand order repaired and checked by reading, not compiled, not run on hardware
//
#include <hip/hip_runtime.h>
#include <math.h>

// ---------------- problem constants (from reference) ----------------
constexpr int  B_     = 2;
constexpr int  N_     = 10000;
constexpr int  F_     = 99;
constexpr int  E_     = 160000;
constexpr int  D_     = 128;
constexpr int  L_     = 4;
constexpr int  NE_    = 64;
constexpr int  DFF_   = 512;
constexpr int  RAIN_START = 17;
constexpr int  RAIN_DIM   = 82;
constexpr int  GH_    = 64;
constexpr long NNT    = (long)B_ * N_;      // 20000 rows
constexpr int  KIN    = F_ + NE_;           // 163
constexpr int  KINP   = 192;                // KIN padded to multiple of 32 (and 8)
constexpr int  NEDGE  = B_ * E_;            // 320000

// ---------------- vector types for WMMA ----------------
typedef __attribute__((ext_vector_type(16))) __bf16        v16bf;
typedef __attribute__((ext_vector_type(8)))  float         v8f;
typedef __attribute__((ext_vector_type(8)))  unsigned int  v8u;

// ---------------- helpers ----------------
__device__ inline unsigned short f2bf(float f) {
    unsigned u = __builtin_bit_cast(unsigned, f);
    unsigned r = u + 0x7FFFu + ((u >> 16) & 1u);   // round-to-nearest-even
    return (unsigned short)(r >> 16);
}
__device__ inline unsigned pk2(float a, float b) {
    return (unsigned)f2bf(a) | ((unsigned)f2bf(b) << 16);
}
__device__ inline float gelu_exact(float v) {
    return 0.5f * v * (1.0f + erff(v * 0.7071067811865476f));
}
__device__ inline float waveRedSum(float v) {
#pragma unroll
    for (int off = 16; off > 0; off >>= 1) v += __shfl_xor(v, off, 32);
    return v;
}
__device__ inline void atomicMaxFloat(float* addr, float value) {
    if (value >= 0.0f)
        atomicMax((int*)addr, __float_as_int(value));
    else
        atomicMin((unsigned int*)addr, __float_as_uint(value));
}

// ---------------- bf16 WMMA GEMM: C = act(A @ W^T + bias) ----------------
// A: (M,K) bf16 row-major, W: (N,K) bf16 row-major, C: (M,N) f32 or bf16.
// Requirements guaranteed by caller: K % 32 == 0, K % 8 == 0, N % 64 == 0.
// Block = 256 threads (8 waves). Block tile 128(M) x 64(N); wave grid 4x2,
// wave tile 32x32 (2x2 WMMA 16x16x32 bf16).  Double-buffered LDS with
// branch-free uint4 register staging (A rows clamped at the M boundary;
// clamped rows only feed discarded C rows).  Epilogue has a block-uniform
// full-tile fast path (no per-element guards).
#define LSTRIDE 40   // bf16 elements per LDS row (80B rows: 16B-aligned)

__device__ inline void stage_global(const unsigned short* __restrict__ A,
                                    const unsigned short* __restrict__ W,
                                    int M, int K,
                                    int m0, int n0, int kk,
                                    int arow, int acol, int brow, int bcol,
                                    uint4* aU, uint4* bU) {
    const int gm = min(m0 + arow, M - 1);          // clamp: safe, rows >= M discarded
    const uint4* ap = (const uint4*)(A + (long)gm * K + kk + acol);
    aU[0] = ap[0];
    aU[1] = ap[1];                                 // 16 bf16
    const int gn = n0 + brow;                      // always < N (N % 64 == 0)
    bU[0] = *(const uint4*)(W + (long)gn * K + kk + bcol);   // 8 bf16
}

__device__ inline void stage_store(unsigned short (*lA)[LSTRIDE],
                                   unsigned short (*lB)[LSTRIDE],
                                   int arow, int acol, int brow, int bcol,
                                   const uint4* aU, const uint4* bU) {
    *(uint4*)&lA[arow][acol]     = aU[0];
    *(uint4*)&lA[arow][acol + 8] = aU[1];
    *(uint4*)&lB[brow][bcol]     = bU[0];
}

__device__ inline v8f wmma_bf16(v16bf a, v16bf b, v8f c) {
    return __builtin_amdgcn_wmma_f32_16x16x32_bf16(false, a, false, b, (short)0, c,
                                                   false, false);
}

__device__ inline void wave_compute(const unsigned short (*lA)[LSTRIDE],
                                    const unsigned short (*lB)[LSTRIDE],
                                    int wm, int wn, int hi, int l15,
                                    v8f acc[2][2]) {
    // preload all fragments, then issue 4 WMMAs back-to-back
    v16bf bfrag[2], afrag[2];
#pragma unroll
    for (int u = 0; u < 2; ++u) {           // B 32x16: col = n, K = hi*16 + e
        v8u bu;
        const int n   = wn + u * 16 + l15;
        const int klo = hi * 16;
#pragma unroll
        for (int vg = 0; vg < 8; ++vg)
            bu[vg] = *(const unsigned*)&lB[n][klo + vg * 2];
        bfrag[u] = __builtin_bit_cast(v16bf, bu);
    }
#pragma unroll
    for (int t = 0; t < 2; ++t) {           // A 16x32: row = m, split-K per half-wave
        v8u au;
        const int m = wm + t * 16 + l15;
#pragma unroll
        for (int vg = 0; vg < 8; ++vg) {
            const int kk = hi * 16 + vg * 2;
            au[vg] = *(const unsigned*)&lA[m][kk];
        }
        afrag[t] = __builtin_bit_cast(v16bf, au);
    }
#pragma unroll
    for (int t = 0; t < 2; ++t) {
        acc[t][0] = wmma_bf16(afrag[t], bfrag[0], acc[t][0]);
        acc[t][1] = wmma_bf16(afrag[t], bfrag[1], acc[t][1]);
    }
}

template <int ACT, int OUTBF>   // ACT: 0 none, 1 exact GELU.  OUTBF: output bf16
__global__ __launch_bounds__(256)
void gemm_bf16_wmma(const unsigned short* __restrict__ A,
                    const unsigned short* __restrict__ W,
                    const float* __restrict__ bias, void* __restrict__ Cv,
                    int M, int N, int K)
{
    __shared__ unsigned short ldsA[2][128][LSTRIDE];
    __shared__ unsigned short ldsB[2][64][LSTRIDE];

    const int tid  = threadIdx.x;
    const int lane = tid & 31;
    const int wave = tid >> 5;           // 0..7
    const int m0   = blockIdx.y * 128;
    const int n0   = blockIdx.x * 64;
    const int wm   = (wave >> 1) * 32;   // wave M offset: 0,32,64,96
    const int wn   = (wave & 1)  * 32;   // wave N offset: 0,32
    const int hi   = lane >> 4;
    const int l15  = lane & 15;

    const int arow = tid >> 1;           // A: half-row (16 bf16) per thread
    const int acol = (tid & 1) * 16;
    const int brow = tid >> 2;           // B: quarter-row (8 bf16) per thread
    const int bcol = (tid & 3) * 8;

    v8f acc[2][2] = {};
    uint4 aU[2], bU[1];

    stage_global(A, W, M, K, m0, n0, 0, arow, acol, brow, bcol, aU, bU);
    stage_store(ldsA[0], ldsB[0], arow, acol, brow, bcol, aU, bU);
    __syncthreads();

    int p = 0;
    for (int k0 = 0; k0 < K; k0 += 32) {
        const bool hasNext = (k0 + 32) < K;
        if (hasNext)
            stage_global(A, W, M, K, m0, n0, k0 + 32, arow, acol, brow, bcol, aU, bU);
        wave_compute(ldsA[p], ldsB[p], wm, wn, hi, l15, acc);
        if (hasNext)
            stage_store(ldsA[p ^ 1], ldsB[p ^ 1], arow, acol, brow, bcol, aU, bU);
        __syncthreads();
        p ^= 1;
    }

    // ---- epilogue: C/D layout -> (M = base + r + hi*8, N = base + l15) ----
    const bool fullM = (m0 + 128 <= M);   // block-uniform fast path
#pragma unroll
    for (int t = 0; t < 2; ++t) {
#pragma unroll
        for (int u = 0; u < 2; ++u) {
            const int n = n0 + wn + u * 16 + l15;
            const float bv = bias[n];
            const int mbase = m0 + wm + t * 16 + hi * 8;
            if (fullM) {
#pragma unroll
                for (int r = 0; r < 8; ++r) {
                    float vv = acc[t][u][r] + bv;
                    if (ACT == 1) vv = gelu_exact(vv);
                    if (OUTBF) ((unsigned short*)Cv)[(long)(mbase + r) * N + n] = f2bf(vv);
                    else       ((float*)Cv)[(long)(mbase + r) * N + n] = vv;
                }
            } else {
#pragma unroll
                for (int r = 0; r < 8; ++r) {
                    if (mbase + r < M) {
                        float vv = acc[t][u][r] + bv;
                        if (ACT == 1) vv = gelu_exact(vv);
                        if (OUTBF) ((unsigned short*)Cv)[(long)(mbase + r) * N + n] = f2bf(vv);
                        else       ((float*)Cv)[(long)(mbase + r) * N + n] = vv;
                    }
                }
            }
        }
    }
}

// ---------------- conversion kernels ----------------
__global__ void cvt_f32_bf16(const float* __restrict__ s,
                             unsigned short* __restrict__ d, long n) {
    const long i = ((long)blockIdx.x * blockDim.x + threadIdx.x) * 2;
    if (i + 1 < n)      *(unsigned*)(d + i) = pk2(s[i], s[i + 1]);
    else if (i < n)     d[i] = f2bf(s[i]);
}

// fp32 (N,K) -> bf16 (N,KP) with zero padding
__global__ void cvt_pad(const float* __restrict__ s, unsigned short* __restrict__ d,
                        int N, int K, int KP) {
    const long i = (long)blockIdx.x * blockDim.x + threadIdx.x;
    if (i >= (long)N * KP) return;
    const int r = (int)(i / KP), c = (int)(i - (long)r * KP);
    d[i] = (c < K) ? f2bf(s[(long)r * K + c]) : (unsigned short)0;
}

// feat = concat(x, node_emb) -> bf16 padded to KINP columns
__global__ void build_feat(const float* __restrict__ x, const float* __restrict__ emb,
                           unsigned short* __restrict__ feat) {
    const long i = (long)blockIdx.x * blockDim.x + threadIdx.x;
    if (i >= NNT * KINP) return;
    const long bn = i / KINP;
    const int  j  = (int)(i - bn * KINP);
    float v = 0.0f;
    if (j < F_)        v = x[bn * F_ + j];
    else if (j < KIN)  v = emb[(long)(bn % N_) * NE_ + (j - F_)];
    feat[i] = f2bf(v);
}

// ---------------- small kernels ----------------
__global__ void fill_f32(float* __restrict__ p, float v, long n) {
    long i = (long)blockIdx.x * blockDim.x + threadIdx.x;
    if (i < n) p[i] = v;
}

__global__ __launch_bounds__(256)
void film_kernel(const float* __restrict__ x,
                 const float* __restrict__ g1w, const float* __restrict__ g1b,
                 const float* __restrict__ g2w, const float* __restrict__ g2b,
                 float* __restrict__ gamma, float* __restrict__ beta) {
    __shared__ float rain[B_][RAIN_DIM];
    __shared__ float gh[B_][GH_];
    const int tid = threadIdx.x;
    if (tid < B_ * RAIN_DIM) {
        const int b = tid / RAIN_DIM, j = tid % RAIN_DIM;
        rain[b][j] = x[(long)b * N_ * F_ + RAIN_START + j];   // x[b,0,17+j]
    }
    __syncthreads();
    if (tid < B_ * GH_) {
        const int b = tid / GH_, i = tid % GH_;
        float acc = g1b[i];
        for (int j = 0; j < RAIN_DIM; ++j) acc += rain[b][j] * g1w[i * RAIN_DIM + j];
        gh[b][i] = gelu_exact(acc);
    }
    __syncthreads();
    for (int idx = tid; idx < B_ * 2 * D_; idx += 256) {
        const int b = idx / (2 * D_), o = idx % (2 * D_);
        float acc = g2b[o];
        for (int j = 0; j < GH_; ++j) acc += gh[b][j] * g2w[o * GH_ + j];
        const float tv = tanhf(acc) * 0.5f;
        if (o < D_) gamma[b * D_ + o]        = tv;
        else        beta [b * D_ + (o - D_)] = tv;
    }
}

__global__ void modulate_kernel(float* __restrict__ h, const float* __restrict__ gamma,
                                const float* __restrict__ beta) {
    long i = (long)blockIdx.x * blockDim.x + threadIdx.x;
    if (i >= NNT * D_) return;
    const int c = (int)(i & (D_ - 1));
    const int b = (int)((i >> 7) / N_);
    h[i] = h[i] * (1.0f + gamma[b * D_ + c]) + beta[b * D_ + c];
}

// wave-per-edge: score = (q[dst] . k[src]) * scale ; atomic segment-max
__global__ __launch_bounds__(256)
void edge_score(const float* __restrict__ q, const float* __restrict__ k,
                const int* __restrict__ ei, float* __restrict__ score,
                float* __restrict__ mbuf) {
    const int g = blockIdx.x * 8 + (threadIdx.x >> 5);
    if (g >= NEDGE) return;
    const int lane = threadIdx.x & 31;
    const int b = g / E_, e = g - b * E_;
    const int src = ei[e]      + b * N_;
    const int dst = ei[E_ + e] + b * N_;
    const float4 qv = *(const float4*)(q + (long)dst * D_ + lane * 4);
    const float4 kv = *(const float4*)(k + (long)src * D_ + lane * 4);
    float p = qv.x * kv.x + qv.y * kv.y + qv.z * kv.z + qv.w * kv.w;
    p = waveRedSum(p);
    if (lane == 0) {
        const float s = p * 0.08838834764831845f;  // D^-0.5
        score[g] = s;
        atomicMaxFloat(mbuf + dst, s);
    }
}

__global__ void edge_exp(float* __restrict__ score, const float* __restrict__ mbuf,
                         float* __restrict__ den, const int* __restrict__ ei) {
    const int g = blockIdx.x * blockDim.x + threadIdx.x;
    if (g >= NEDGE) return;
    const int b = g / E_, e = g - b * E_;
    const int dst = ei[E_ + e] + b * N_;
    const float ew = expf(score[g] - mbuf[dst]);
    score[g] = ew;
    atomicAdd(den + dst, ew);
}

// wave-per-edge: agg[dst] += v[src] * alpha
__global__ __launch_bounds__(256)
void edge_agg(const float* __restrict__ v, const int* __restrict__ ei,
              const float* __restrict__ ew, const float* __restrict__ den,
              float* __restrict__ agg) {
    const int g = blockIdx.x * 8 + (threadIdx.x >> 5);
    if (g >= NEDGE) return;
    const int lane = threadIdx.x & 31;
    const int b = g / E_, e = g - b * E_;
    const int src = ei[e]      + b * N_;
    const int dst = ei[E_ + e] + b * N_;
    const float alpha = ew[g] / fmaxf(den[dst], 1e-12f);
    const float4 vv = *(const float4*)(v + (long)src * D_ + lane * 4);
    float* ap = agg + (long)dst * D_ + lane * 4;
    atomicAdd(ap + 0, vv.x * alpha);
    atomicAdd(ap + 1, vv.y * alpha);
    atomicAdd(ap + 2, vv.z * alpha);
    atomicAdd(ap + 3, vv.w * alpha);
}

// wave-per-row residual + LayerNorm.  MODE 0: t = h + relu(a + b2);  MODE 1: t = h + a
template <int MODE>
__global__ __launch_bounds__(256)
void ln_residual(float* __restrict__ h, const float* __restrict__ a,
                 const float* __restrict__ b2, const float* __restrict__ sc,
                 const float* __restrict__ bi, int M) {
    const int row = blockIdx.x * 8 + (threadIdx.x >> 5);
    if (row >= M) return;
    const int lane = threadIdx.x & 31;
    float* hp = h + (long)row * D_;
    const float4 hv = *(const float4*)(hp + lane * 4);
    float4 t;
    if (MODE == 0) {
        const float4 sv = *(const float4*)(a  + (long)row * D_ + lane * 4);
        const float4 av = *(const float4*)(b2 + (long)row * D_ + lane * 4);
        t.x = hv.x + fmaxf(sv.x + av.x, 0.0f);
        t.y = hv.y + fmaxf(sv.y + av.y, 0.0f);
        t.z = hv.z + fmaxf(sv.z + av.z, 0.0f);
        t.w = hv.w + fmaxf(sv.w + av.w, 0.0f);
    } else {
        const float4 fv = *(const float4*)(a + (long)row * D_ + lane * 4);
        t.x = hv.x + fv.x; t.y = hv.y + fv.y; t.z = hv.z + fv.z; t.w = hv.w + fv.w;
    }
    const float mean = waveRedSum(t.x + t.y + t.z + t.w) * (1.0f / D_);
    const float dx = t.x - mean, dy = t.y - mean, dz = t.z - mean, dw = t.w - mean;
    const float var = waveRedSum(dx * dx + dy * dy + dz * dz + dw * dw) * (1.0f / D_);
    const float inv = rsqrtf(var + 1e-5f);
    float4 o;
    o.x = dx * inv * sc[lane * 4 + 0] + bi[lane * 4 + 0];
    o.y = dy * inv * sc[lane * 4 + 1] + bi[lane * 4 + 1];
    o.z = dz * inv * sc[lane * 4 + 2] + bi[lane * 4 + 2];
    o.w = dw * inv * sc[lane * 4 + 3] + bi[lane * 4 + 3];
    *(float4*)(hp + lane * 4) = o;
}

// wave-per-row: LN(h) then 2-way head projection
__global__ __launch_bounds__(256)
void head_kernel(const float* __restrict__ h, const float* __restrict__ hs,
                 const float* __restrict__ hb, const float* __restrict__ hw,
                 const float* __restrict__ hbias, float* __restrict__ out, int M) {
    const int row = blockIdx.x * 8 + (threadIdx.x >> 5);
    if (row >= M) return;
    const int lane = threadIdx.x & 31;
    const float4 hv = *(const float4*)(h + (long)row * D_ + lane * 4);
    const float mean = waveRedSum(hv.x + hv.y + hv.z + hv.w) * (1.0f / D_);
    const float dx = hv.x - mean, dy = hv.y - mean, dz = hv.z - mean, dw = hv.w - mean;
    const float var = waveRedSum(dx * dx + dy * dy + dz * dz + dw * dw) * (1.0f / D_);
    const float inv = rsqrtf(var + 1e-5f);
    const float nx = dx * inv * hs[lane * 4 + 0] + hb[lane * 4 + 0];
    const float ny = dy * inv * hs[lane * 4 + 1] + hb[lane * 4 + 1];
    const float nz = dz * inv * hs[lane * 4 + 2] + hb[lane * 4 + 2];
    const float nw = dw * inv * hs[lane * 4 + 3] + hb[lane * 4 + 3];
    const float4 w0 = *(const float4*)(hw +      lane * 4);
    const float4 w1 = *(const float4*)(hw + D_ + lane * 4);
    const float d0 = waveRedSum(nx * w0.x + ny * w0.y + nz * w0.z + nw * w0.w);
    const float d1 = waveRedSum(nx * w1.x + ny * w1.y + nz * w1.z + nw * w1.w);
    if (lane == 0) {
        out[row]     = d0 + hbias[0];
        out[M + row] = d1 + hbias[1];
    }
}

// ---------------- host orchestration ----------------
static inline dim3 gemm_grid(int M, int N) { return dim3((N + 63) / 64, (M + 127) / 128); }
static inline unsigned gpairs(long n) { return (unsigned)((n / 2 + 255) / 256); }

extern "C" void kernel_launch(void* const* d_in, const int* in_sizes, int n_in,
                              void* d_out, int out_size, void* d_ws, size_t ws_size,
                              hipStream_t stream) {
    (void)in_sizes; (void)n_in; (void)out_size; (void)ws_size;

    const float* x      = (const float*)d_in[0];
    const int*   ei     = (const int*)  d_in[1];
    const float* emb    = (const float*)d_in[2];
    const float* W_in   = (const float*)d_in[3];
    const float* b_in   = (const float*)d_in[4];
    const float* g1w    = (const float*)d_in[5];
    const float* g1b    = (const float*)d_in[6];
    const float* g2w    = (const float*)d_in[7];
    const float* g2b    = (const float*)d_in[8];
    const float* qw     = (const float*)d_in[9];
    const float* kw     = (const float*)d_in[10];
    const float* vw     = (const float*)d_in[11];
    const float* sw     = (const float*)d_in[12];
    const float* ow     = (const float*)d_in[13];
    const float* qb     = (const float*)d_in[14];
    const float* kb     = (const float*)d_in[15];
    const float* vb     = (const float*)d_in[16];
    const float* sb     = (const float*)d_in[17];
    const float* ob     = (const float*)d_in[18];
    const float* n1s    = (const float*)d_in[19];
    const float* n1b    = (const float*)d_in[20];
    const float* f1w    = (const float*)d_in[21];
    const float* f1b    = (const float*)d_in[22];
    const float* f2w    = (const float*)d_in[23];
    const float* f2b    = (const float*)d_in[24];
    const float* n2s    = (const float*)d_in[25];
    const float* n2b    = (const float*)d_in[26];
    const float* hs     = (const float*)d_in[27];
    const float* hb     = (const float*)d_in[28];
    const float* head_w = (const float*)d_in[29];
    const float* head_b = (const float*)d_in[30];
    float* out = (float*)d_out;

    // ---- workspace layout (byte allocator, 16B aligned) ----
    char*  base = (char*)d_ws;
    size_t off  = 0;
    auto alloc = [&](size_t bytes) -> void* {
        void* p = base + off;
        off += (bytes + 15) & ~(size_t)15;
        return p;
    };
    float* bufH   = (float*)alloc(NNT * D_   * 4);
    float* bufQ   = (float*)alloc(NNT * D_   * 4);   // reused as agg
    float* bufK   = (float*)alloc(NNT * D_   * 4);   // reused as agg@ow^T
    float* bufV   = (float*)alloc(NNT * D_   * 4);   // reused as ff2
    float* bufS   = (float*)alloc(NNT * D_   * 4);
    float* bufE   = (float*)alloc((long)NEDGE * 4);
    float* bufM   = (float*)alloc(NNT * 4);
    float* bufDen = (float*)alloc(NNT * 4);
    float* gamma  = (float*)alloc(B_ * D_ * 4);
    float* beta   = (float*)alloc(B_ * D_ * 4);
    unsigned short* hB    = (unsigned short*)alloc(NNT * D_   * 2);
    unsigned short* aggB  = (unsigned short*)alloc(NNT * D_   * 2);
    unsigned short* ffB   = (unsigned short*)alloc(NNT * DFF_ * 2);
    unsigned short* featB = (unsigned short*)alloc(NNT * KINP * 2);
    unsigned short* wInB  = (unsigned short*)alloc((long)D_ * KINP * 2);
    unsigned short* qwB   = (unsigned short*)alloc((long)L_ * D_ * D_ * 2);
    unsigned short* kwB   = (unsigned short*)alloc((long)L_ * D_ * D_ * 2);
    unsigned short* vwB   = (unsigned short*)alloc((long)L_ * D_ * D_ * 2);
    unsigned short* swB   = (unsigned short*)alloc((long)L_ * D_ * D_ * 2);
    unsigned short* owB   = (unsigned short*)alloc((long)L_ * D_ * D_ * 2);
    unsigned short* f1wB  = (unsigned short*)alloc((long)L_ * DFF_ * D_ * 2);
    unsigned short* f2wB  = (unsigned short*)alloc((long)L_ * D_ * DFF_ * 2);

    const int  M  = (int)NNT;
    const int  TB = 256;
    const long WDD = (long)L_ * D_ * D_;
    const long WF  = (long)L_ * DFF_ * D_;

    // ---- 0) convert all weights to bf16 once ----
    cvt_pad<<<(unsigned)(((long)D_ * KINP + TB - 1) / TB), TB, 0, stream>>>(W_in, wInB, D_, KIN, KINP);
    cvt_f32_bf16<<<gpairs(WDD), TB, 0, stream>>>(qw, qwB, WDD);
    cvt_f32_bf16<<<gpairs(WDD), TB, 0, stream>>>(kw, kwB, WDD);
    cvt_f32_bf16<<<gpairs(WDD), TB, 0, stream>>>(vw, vwB, WDD);
    cvt_f32_bf16<<<gpairs(WDD), TB, 0, stream>>>(sw, swB, WDD);
    cvt_f32_bf16<<<gpairs(WDD), TB, 0, stream>>>(ow, owB, WDD);
    cvt_f32_bf16<<<gpairs(WF),  TB, 0, stream>>>(f1w, f1wB, WF);
    cvt_f32_bf16<<<gpairs(WF),  TB, 0, stream>>>(f2w, f2wB, WF);

    // ---- 1) feat (bf16, K padded) ; h = feat @ W_in^T + b_in ----
    build_feat<<<(unsigned)((NNT * KINP + TB - 1) / TB), TB, 0, stream>>>(x, emb, featB);
    gemm_bf16_wmma<0, 0><<<gemm_grid(M, D_), 256, 0, stream>>>(featB, wInB, b_in, bufH, M, D_, KINP);

    // ---- 2) FiLM modulation ----
    film_kernel<<<1, 256, 0, stream>>>(x, g1w, g1b, g2w, g2b, gamma, beta);
    modulate_kernel<<<(unsigned)((NNT * D_ + TB - 1) / TB), TB, 0, stream>>>(bufH, gamma, beta);

    // ---- 3) transformer layers ----
    for (int i = 0; i < L_; ++i) {
        const long wo = (long)i * D_ * D_;
        const long bo = (long)i * D_;

        cvt_f32_bf16<<<gpairs(NNT * D_), TB, 0, stream>>>(bufH, hB, NNT * D_);
        gemm_bf16_wmma<0, 0><<<gemm_grid(M, D_), 256, 0, stream>>>(hB, qwB + wo, qb + bo, bufQ, M, D_, D_);
        gemm_bf16_wmma<0, 0><<<gemm_grid(M, D_), 256, 0, stream>>>(hB, kwB + wo, kb + bo, bufK, M, D_, D_);
        gemm_bf16_wmma<0, 0><<<gemm_grid(M, D_), 256, 0, stream>>>(hB, vwB + wo, vb + bo, bufV, M, D_, D_);
        gemm_bf16_wmma<0, 0><<<gemm_grid(M, D_), 256, 0, stream>>>(hB, swB + wo, sb + bo, bufS, M, D_, D_);

        fill_f32<<<(unsigned)((NNT + TB - 1) / TB), TB, 0, stream>>>(bufM, -3.0e38f, NNT);
        fill_f32<<<(unsigned)((NNT + TB - 1) / TB), TB, 0, stream>>>(bufDen, 0.0f, NNT);
        edge_score<<<(NEDGE + 7) / 8, 256, 0, stream>>>(bufQ, bufK, ei, bufE, bufM);
        fill_f32<<<(unsigned)((NNT * D_ + TB - 1) / TB), TB, 0, stream>>>(bufQ, 0.0f, NNT * D_); // agg
        edge_exp<<<(NEDGE + TB - 1) / TB, TB, 0, stream>>>(bufE, bufM, bufDen, ei);
        edge_agg<<<(NEDGE + 7) / 8, 256, 0, stream>>>(bufV, ei, bufE, bufDen, bufQ);

        cvt_f32_bf16<<<gpairs(NNT * D_), TB, 0, stream>>>(bufQ, aggB, NNT * D_);
        gemm_bf16_wmma<0, 0><<<gemm_grid(M, D_), 256, 0, stream>>>(aggB, owB + wo, ob + bo, bufK, M, D_, D_);
        ln_residual<0><<<(M + 7) / 8, 256, 0, stream>>>(bufH, bufS, bufK, n1s + bo, n1b + bo, M);

        cvt_f32_bf16<<<gpairs(NNT * D_), TB, 0, stream>>>(bufH, hB, NNT * D_);
        gemm_bf16_wmma<1, 1><<<gemm_grid(M, DFF_), 256, 0, stream>>>(hB, f1wB + (long)i * DFF_ * D_,
                                                                     f1b + (long)i * DFF_, ffB, M, DFF_, D_);
        gemm_bf16_wmma<0, 0><<<gemm_grid(M, D_), 256, 0, stream>>>(ffB, f2wB + (long)i * D_ * DFF_,
                                                                   f2b + bo, bufV, M, D_, DFF_);
        ln_residual<1><<<(M + 7) / 8, 256, 0, stream>>>(bufH, bufV, nullptr, n2s + bo, n2b + bo, M);
    }

    // ---- 4) head ----
    head_kernel<<<(M + 7) / 8, 256, 0, stream>>>(bufH, hs, hb, head_w, head_b, out, M);
}
